// CrossGraphAttention_67997922230508
// MI455X (gfx1250) — matrix operand order repaired and checked by reading, compiled, not run on hardware
//
#include <hip/hip_runtime.h>
#include <hip/hip_bf16.h>

// ---------------------------------------------------------------------------
// CrossGraphAttention fused CDNA5 (gfx1250) implementation.
// Shapes (fixed by reference): B=2, L=512, dn=128, dE=64, dD=64, H=8,
// dk=16, de=8, dd=8.
// ---------------------------------------------------------------------------

typedef __attribute__((ext_vector_type(16))) _Float16 v16h;
typedef __attribute__((ext_vector_type(8)))  _Float16 v8h;
typedef __attribute__((ext_vector_type(8)))  float    v8f;

#define NB 2
#define LL 512
#define DN 128
#define DEE 64
#define DDD 64
#define HH 8

// ---- workspace layout (float offsets) ----
#define OFF_B1     0u
#define OFF_T2     131072u
#define OFF_ROW    262144u
#define OFF_COL    327680u
#define OFF_HJW    393216u
#define OFF_TJW    524288u
#define OFF_SCI_T  655360u
#define OFF_SCJ_T  663552u
#define OFF_SCI_B  671744u
#define OFF_SCJ_B  679936u
#define OFF_BIAS1  688128u
#define OFF_NEWT   688192u
#define OFF_NEWB   819264u
#define OFF_SET    950336u
#define OFF_SEB    5144640u
#define OFF_HALF   9338944u   // byte offset 37355776, 32B aligned; halves after

__device__ __forceinline__ float lrelu(float x) { return x >= 0.f ? x : 0.2f * x; }

__device__ __forceinline__ v8f wmma16(v16h a, v16h b, v8f c) {
  // (neg_a, A, neg_b, B, c_mod, C, reuse_a, reuse_b)
  return __builtin_amdgcn_wmma_f32_16x16x32_f16(false, a, false, b, (short)0, c,
                                                false, false);
}

__device__ __forceinline__ v16h pack16(float4 f0, float4 f1, float4 f2, float4 f3) {
  v16h a;
  a[0]=(_Float16)f0.x; a[1]=(_Float16)f0.y; a[2]=(_Float16)f0.z; a[3]=(_Float16)f0.w;
  a[4]=(_Float16)f1.x; a[5]=(_Float16)f1.y; a[6]=(_Float16)f1.z; a[7]=(_Float16)f1.w;
  a[8]=(_Float16)f2.x; a[9]=(_Float16)f2.y; a[10]=(_Float16)f2.z; a[11]=(_Float16)f2.w;
  a[12]=(_Float16)f3.x; a[13]=(_Float16)f3.y; a[14]=(_Float16)f3.z; a[15]=(_Float16)f3.w;
  return a;
}

// ---------------------------------------------------------------------------
// P0: fold We_w * blockdiag_H(w_e) -> M1 ; transpose weights to f16 B-layout
// ---------------------------------------------------------------------------
__global__ void weight_prep(const float* __restrict__ We_w,
                            const float* __restrict__ We_b,
                            const float* __restrict__ el_w,
                            const float* __restrict__ ope_w,
                            const float* __restrict__ We2_w,
                            _Float16* __restrict__ M1T,
                            _Float16* __restrict__ opeT,
                            _Float16* __restrict__ We2T,
                            float* __restrict__ bias1) {
  int t = threadIdx.x;
  for (int idx = t; idx < 64 * 64; idx += 256) {
    int c = idx >> 6, k = idx & 63;      // c = output channel (8h+f), k = input
    int h = c >> 3, f = c & 7;
    float s = 0.f;
#pragma unroll
    for (int e = 0; e < 8; ++e) s += We_w[k * 64 + h * 8 + e] * el_w[e * 8 + f];
    M1T[c * 64 + k]  = (_Float16)s;                 // transposed (N-major)
    opeT[c * 64 + k] = (_Float16)ope_w[k * 64 + c];
    We2T[c * 64 + k] = (_Float16)We2_w[k * 64 + c];
  }
  if (t < 64) {
    int h = t >> 3, f = t & 7;
    float s = 0.f;
#pragma unroll
    for (int e = 0; e < 8; ++e) s += We_b[h * 8 + e] * el_w[e * 8 + f];
    bias1[t] = s;
  }
}

// ---------------------------------------------------------------------------
// P1: all node-level projections + broadcast terms + score partials
// grid = B*L blocks, 128 threads
// ---------------------------------------------------------------------------
__global__ void node_prep(const float* __restrict__ target,
                          const float* __restrict__ binder,
                          const float* __restrict__ diff,
                          const float* __restrict__ Wn_w, const float* __restrict__ Wn_b,
                          const float* __restrict__ Wn2_w, const float* __restrict__ Wn2_b,
                          const float* __restrict__ Wd_w, const float* __restrict__ Wd_b,
                          const float* __restrict__ Wd2_w, const float* __restrict__ Wd2_b,
                          const float* __restrict__ Wh_w, const float* __restrict__ Wh_b,
                          const float* __restrict__ el_w, const float* __restrict__ el_b,
                          const float* __restrict__ attn_t, const float* __restrict__ attn_bind,
                          const float* __restrict__ bias1,
                          float* __restrict__ b1o, float* __restrict__ t2o,
                          float* __restrict__ rowterm, float* __restrict__ colterm,
                          float* __restrict__ hjW, float* __restrict__ tjW,
                          float* __restrict__ sci_t, float* __restrict__ scj_t,
                          float* __restrict__ sci_b, float* __restrict__ scj_b) {
  int m = blockIdx.x;      // flat node (b*L + x)
  int c = threadIdx.x;     // 0..127
  __shared__ float xt[128], xb[128], xd[64];
  __shared__ float t1[128], b1[128], t2[128], d1[64], d2[64];
  xt[c] = target[m * 128 + c];
  xb[c] = binder[m * 128 + c];
  if (c < 64) xd[c] = diff[m * 64 + c];
  __syncthreads();

  float s_t1 = Wn_b[c], s_b1 = Wn_b[c], s_t2 = Wn2_b[c];
  for (int k = 0; k < 128; ++k) {
    float w1 = Wn_w[k * 128 + c], w2 = Wn2_w[k * 128 + c];
    s_t1 += xt[k] * w1; s_b1 += xb[k] * w1; s_t2 += xt[k] * w2;
  }
  t1[c] = s_t1; b1[c] = s_b1; t2[c] = s_t2;
  if (c < 64) {
    float sd1 = Wd_b[c], sd2 = Wd2_b[c];
    for (int k = 0; k < 64; ++k) {
      sd1 += xd[k] * Wd_w[k * 64 + c];
      sd2 += xd[k] * Wd2_w[k * 64 + c];
    }
    d1[c] = sd1; d2[c] = sd2;
  }
  __syncthreads();

  b1o[m * 128 + c] = b1[c];
  t2o[m * 128 + c] = t2[c];

  if (c < 64) {   // edge-update broadcast terms per channel c = 8h+f
    int h = c >> 3, f = c & 7;
    float rt = el_b[f] + bias1[c], ct = 0.f;
#pragma unroll
    for (int d = 0; d < 16; ++d) {
      rt += t1[h * 16 + d] * el_w[(8 + d) * 8 + f];    // w_hi
      ct += b1[h * 16 + d] * el_w[(24 + d) * 8 + f];   // w_hj
    }
#pragma unroll
    for (int d = 0; d < 8; ++d) rt += d1[h * 8 + d] * el_w[(40 + d) * 8 + f];  // w_d
    rowterm[m * 64 + c] = rt;
    colterm[m * 64 + c] = ct;
  }
  {   // hjW (from b1) and tjW (from t2): per-head 16x16 with Wh
    int h = c >> 4, fo = c & 15;
    float hw = Wh_b[fo], tw = Wh_b[fo];
#pragma unroll
    for (int d = 0; d < 16; ++d) {
      float w = Wh_w[d * 16 + fo];
      hw += b1[h * 16 + d] * w;
      tw += t2[h * 16 + d] * w;
    }
    hjW[m * 128 + c] = hw;
    tjW[m * 128 + c] = tw;
  }
  if (c < 8) {    // attention score node partials
    int h = c;
    float v1 = 0, v2 = 0, v3 = 0, v4 = 0, v5 = 0;
#pragma unroll
    for (int d = 0; d < 16; ++d) {
      v1 += t2[h * 16 + d] * attn_t[d];
      v2 += b1[h * 16 + d] * attn_t[16 + d];
      v4 += b1[h * 16 + d] * attn_bind[d];
      v5 += t2[h * 16 + d] * attn_bind[16 + d];
    }
#pragma unroll
    for (int d = 0; d < 8; ++d) v3 += d2[h * 8 + d] * attn_t[40 + d];
    sci_t[m * 8 + h] = v1 + v3;
    scj_t[m * 8 + h] = v2;
    sci_b[m * 8 + h] = v4;
    scj_b[m * 8 + h] = v5;
  }
}

// ---------------------------------------------------------------------------
// Mega kernel: one pass over the pair tensor.
// For each (b,i): edge row (512x64) -> GEMM1(+bcast,+diag0) -> GEMM2(ope)
//   -> write new_edge -> GEMM3(We2) -> reduce e2 to per-(i,j,h) score scalars.
// grid = B*L blocks, 256 threads (8 waves); each wave owns 16x64 tiles.
// ---------------------------------------------------------------------------
__global__ void __launch_bounds__(256) edge_mega(
    const float* __restrict__ edge,
    const _Float16* __restrict__ M1T, const _Float16* __restrict__ opeT,
    const _Float16* __restrict__ We2T,
    const float* __restrict__ rowterm, const float* __restrict__ colterm,
    const float* __restrict__ ope_b, const float* __restrict__ We2_b,
    const float* __restrict__ attn_t, const float* __restrict__ attn_bind,
    float* __restrict__ new_edge, float* __restrict__ sE_t,
    float* __restrict__ sE_b) {
  const int blk = blockIdx.x, b = blk >> 9, i = blk & 511;
  const int tid = threadIdx.x, wv = tid >> 5, lane = tid & 31;
  const int n16 = lane & 15, g = lane >> 4;

  // resident B fragments (weights, f16, B-layout: lane=N col, 16 contiguous K)
  v16h W1[4][2], W2[4][2], W3[4][2];
#pragma unroll
  for (int nb = 0; nb < 4; ++nb)
#pragma unroll
    for (int ks = 0; ks < 2; ++ks) {
      int off = (nb * 16 + n16) * 64 + ks * 32 + g * 16;
      W1[nb][ks] = *(const v16h*)(M1T + off);
      W2[nb][ks] = *(const v16h*)(opeT + off);
      W3[nb][ks] = *(const v16h*)(We2T + off);
    }
  float rt[4], ob[4], wb[4];
#pragma unroll
  for (int nb = 0; nb < 4; ++nb) {
    int c = nb * 16 + n16;
    rt[nb] = rowterm[(b * 512 + i) * 64 + c];
    ob[nb] = ope_b[c];
    wb[nb] = We2_b[c];
  }
  const float ae  = attn_t[32 + (n16 & 7)];
  const float abe = attn_bind[32 + (n16 & 7)];

  __shared__ _Float16 xbuf[8][16 * 72];   // wave-private 16x64 tile (+pad)
  _Float16* xt = &xbuf[wv][0];

  const float* erow = edge + ((size_t)(b * 512 + i) * 512) * 64;
  float* orow = new_edge + ((size_t)(b * 512 + i) * 512) * 64;

  for (int jt = wv; jt < 32; jt += 8) {
    const int jb = jt * 16;
    if (jt + 8 < 32)   // stream-prefetch next tile for this wave
      __builtin_prefetch(erow + (size_t)(jb + 128) * 64 + lane * 32, 0, 1);

    // ---- GEMM1: new_e = edge @ M1 (+row/col terms, diag->0) ----
    v8f acc[4];
#pragma unroll
    for (int nb = 0; nb < 4; ++nb) acc[nb] = (v8f){0,0,0,0,0,0,0,0};
#pragma unroll
    for (int ks = 0; ks < 2; ++ks) {
      const float* p = erow + (size_t)(jb + n16) * 64 + ks * 32 + g * 16;
      v16h A = pack16(*(const float4*)p, *(const float4*)(p + 4),
                      *(const float4*)(p + 8), *(const float4*)(p + 12));
#pragma unroll
      for (int nb = 0; nb < 4; ++nb) acc[nb] = wmma16(A, W1[nb][ks], acc[nb]);
    }
#pragma unroll
    for (int nb = 0; nb < 4; ++nb) {
      int c = nb * 16 + n16;
#pragma unroll
      for (int r = 0; r < 8; ++r) {
        int jrow = jb + r + 8 * g;
        float v = acc[nb][r] + rt[nb] + colterm[(size_t)(b * 512 + jrow) * 64 + c];
        if (jrow == i) v = 0.f;
        xt[(r + 8 * g) * 72 + c] = (_Float16)v;
      }
    }

    // ---- GEMM2: new_edge = new_e @ ope_w + ope_b ----
    v8f acc2[4];
#pragma unroll
    for (int nb = 0; nb < 4; ++nb) acc2[nb] = (v8f){0,0,0,0,0,0,0,0};
#pragma unroll
    for (int ks = 0; ks < 2; ++ks) {
      const _Float16* p = xt + n16 * 72 + ks * 32 + g * 16;
      union { v16h v; v8h h8[2]; } u;
      u.h8[0] = *(const v8h*)p;
      u.h8[1] = *(const v8h*)(p + 8);
#pragma unroll
      for (int nb = 0; nb < 4; ++nb) acc2[nb] = wmma16(u.v, W2[nb][ks], acc2[nb]);
    }
#pragma unroll
    for (int nb = 0; nb < 4; ++nb) {
      int c = nb * 16 + n16;
#pragma unroll
      for (int r = 0; r < 8; ++r) {
        int jrow = jb + r + 8 * g;
        float v = acc2[nb][r] + ob[nb];
        orow[(size_t)jrow * 64 + c] = v;           // output new_edge
        xt[(r + 8 * g) * 72 + c] = (_Float16)v;    // feed GEMM3
      }
    }

    // ---- GEMM3: e2 = new_edge @ We2_w + We2_b, reduce to score scalars ----
    v8f acc3[4];
#pragma unroll
    for (int nb = 0; nb < 4; ++nb) acc3[nb] = (v8f){0,0,0,0,0,0,0,0};
#pragma unroll
    for (int ks = 0; ks < 2; ++ks) {
      const _Float16* p = xt + n16 * 72 + ks * 32 + g * 16;
      union { v16h v; v8h h8[2]; } u;
      u.h8[0] = *(const v8h*)p;
      u.h8[1] = *(const v8h*)(p + 8);
#pragma unroll
      for (int nb = 0; nb < 4; ++nb) acc3[nb] = wmma16(u.v, W3[nb][ks], acc3[nb]);
    }
#pragma unroll
    for (int nb = 0; nb < 4; ++nb) {
#pragma unroll
      for (int r = 0; r < 8; ++r) {
        float z = acc3[nb][r] + wb[nb];
        float vt = z * ae, vb = z * abe;
        vt += __shfl_xor(vt, 1); vt += __shfl_xor(vt, 2); vt += __shfl_xor(vt, 4);
        vb += __shfl_xor(vb, 1); vb += __shfl_xor(vb, 2); vb += __shfl_xor(vb, 4);
        if ((lane & 7) == 0) {
          int jrow = jb + r + 8 * g;
          int h = (nb * 16 + n16) >> 3;
          sE_t[((size_t)(b * 512 + i) * 512 + jrow) * 8 + h] = vt;
          sE_b[((size_t)(b * 512 + jrow) * 512 + i) * 8 + h] = vb;   // pre-transposed
        }
      }
    }
  }
}

// ---------------------------------------------------------------------------
// Attention (shared by target/binder): softmax over j + weighted sum of W rows
// grid = B*L blocks, 256 threads; wave w = head h.
// ---------------------------------------------------------------------------
__global__ void __launch_bounds__(256) attn_kernel(
    const float* __restrict__ sci, const float* __restrict__ scj,
    const float* __restrict__ sE, const float* __restrict__ Wrow,
    const float* __restrict__ res, float* __restrict__ outbuf) {
  int blk = blockIdx.x, b = blk >> 9, i = blk & 511;
  int h = threadIdx.x >> 5, l = threadIdx.x & 31;
  __shared__ float pbuf[8][512];

  float ci = sci[(b * 512 + i) * 8 + h];
  size_t base = ((size_t)(b * 512 + i) * 512) * 8 + h;
  float mx = -3.4e38f;
  float sv[16];
#pragma unroll
  for (int t = 0; t < 16; ++t) {
    int j = t * 32 + l;
    float s = ci + scj[(b * 512 + j) * 8 + h] + sE[base + (size_t)j * 8];
    s = lrelu(s);
    if (j == i) s = -1e9f;
    sv[t] = s;
    mx = fmaxf(mx, s);
  }
  for (int d = 1; d < 32; d <<= 1) mx = fmaxf(mx, __shfl_xor(mx, d));
  float sum = 0.f;
#pragma unroll
  for (int t = 0; t < 16; ++t) {
    int j = t * 32 + l;
    float p = __expf(sv[t] - mx);
    sum += p;
    pbuf[h][j] = p;
  }
  for (int d = 1; d < 32; d <<= 1) sum += __shfl_xor(sum, d);
  __syncthreads();

  int f = l & 15, jh = l >> 4;
  float acc = 0.f;
  for (int j = jh * 256; j < jh * 256 + 256; ++j)
    acc += pbuf[h][j] * Wrow[((size_t)(b * 512 + j) * 8 + h) * 16 + f];
  acc += __shfl_xor(acc, 16);
  if (l < 16) {
    size_t o = (size_t)(b * 512 + i) * 128 + h * 16 + f;
    outbuf[o] = res[o] + lrelu(acc / sum);
  }
}

// ---------------------------------------------------------------------------
// Final node output projection: out = x @ opn_w + opn_b  (rows 0..1023 = new_t,
// 1024..2047 = new_b) -> contiguous (out_t, out_b) in d_out.
// ---------------------------------------------------------------------------
__global__ void outproj(const float* __restrict__ new_t,
                        const float* __restrict__ new_b,
                        const float* __restrict__ opn_w,
                        const float* __restrict__ opn_b,
                        float* __restrict__ out) {
  int r = blockIdx.x;
  const float* src = (r < 1024) ? (new_t + (size_t)r * 128)
                                : (new_b + (size_t)(r - 1024) * 128);
  __shared__ float x[128];
  int c = threadIdx.x;
  x[c] = src[c];
  __syncthreads();
  float acc = opn_b[c];
  for (int k = 0; k < 128; ++k) acc += x[k] * opn_w[k * 128 + c];
  out[(size_t)r * 128 + c] = acc;
}

// ---------------------------------------------------------------------------
extern "C" void kernel_launch(void* const* d_in, const int* in_sizes, int n_in,
                              void* d_out, int out_size, void* d_ws, size_t ws_size,
                              hipStream_t stream) {
  const float* target   = (const float*)d_in[0];
  const float* binder   = (const float*)d_in[1];
  const float* edge     = (const float*)d_in[2];
  const float* diff     = (const float*)d_in[3];
  const float* Wn_w     = (const float*)d_in[4];
  const float* Wn_b     = (const float*)d_in[5];
  const float* Wh_w     = (const float*)d_in[6];
  const float* Wh_b     = (const float*)d_in[7];
  const float* We_w     = (const float*)d_in[8];
  const float* We_b     = (const float*)d_in[9];
  const float* Wn2_w    = (const float*)d_in[10];
  const float* Wn2_b    = (const float*)d_in[11];
  const float* We2_w    = (const float*)d_in[12];
  const float* We2_b    = (const float*)d_in[13];
  const float* Wd_w     = (const float*)d_in[14];
  const float* Wd_b     = (const float*)d_in[15];
  const float* Wd2_w    = (const float*)d_in[16];
  const float* Wd2_b    = (const float*)d_in[17];
  const float* attn_t   = (const float*)d_in[18];
  const float* attn_bd  = (const float*)d_in[19];
  const float* el_w     = (const float*)d_in[20];
  const float* el_b     = (const float*)d_in[21];
  const float* opn_w    = (const float*)d_in[22];
  const float* opn_b    = (const float*)d_in[23];
  const float* ope_w    = (const float*)d_in[24];
  const float* ope_b    = (const float*)d_in[25];

  float* ws = (float*)d_ws;
  _Float16* halfbase = (_Float16*)((char*)d_ws + (size_t)OFF_HALF * 4);
  _Float16* M1T  = halfbase;
  _Float16* opeT = halfbase + 4096;
  _Float16* We2T = halfbase + 8192;

  float* out_nodes = (float*)d_out;              // out_t then out_b (2*131072)
  float* out_edge  = (float*)d_out + 262144;     // new_edge

  weight_prep<<<1, 256, 0, stream>>>(We_w, We_b, el_w, ope_w, We2_w,
                                     M1T, opeT, We2T, ws + OFF_BIAS1);

  node_prep<<<NB * LL, 128, 0, stream>>>(
      target, binder, diff, Wn_w, Wn_b, Wn2_w, Wn2_b, Wd_w, Wd_b, Wd2_w, Wd2_b,
      Wh_w, Wh_b, el_w, el_b, attn_t, attn_bd, ws + OFF_BIAS1,
      ws + OFF_B1, ws + OFF_T2, ws + OFF_ROW, ws + OFF_COL,
      ws + OFF_HJW, ws + OFF_TJW,
      ws + OFF_SCI_T, ws + OFF_SCJ_T, ws + OFF_SCI_B, ws + OFF_SCJ_B);

  edge_mega<<<NB * LL, 256, 0, stream>>>(
      edge, M1T, opeT, We2T, ws + OFF_ROW, ws + OFF_COL, ope_b, We2_b,
      attn_t, attn_bd, out_edge, ws + OFF_SET, ws + OFF_SEB);

  attn_kernel<<<NB * LL, 256, 0, stream>>>(ws + OFF_SCI_T, ws + OFF_SCJ_T,
                                           ws + OFF_SET, ws + OFF_HJW,
                                           ws + OFF_T2, ws + OFF_NEWT);
  attn_kernel<<<NB * LL, 256, 0, stream>>>(ws + OFF_SCI_B, ws + OFF_SCJ_B,
                                           ws + OFF_SEB, ws + OFF_TJW,
                                           ws + OFF_B1, ws + OFF_NEWB);

  outproj<<<2 * NB * LL, 128, 0, stream>>>(ws + OFF_NEWT, ws + OFF_NEWB,
                                           opn_w, opn_b, out_nodes);
}
